// ECOConv_62526133895816
// MI455X (gfx1250) — compile-verified, matrix operand order checked by reading, not run on hardware
//
#include <hip/hip_runtime.h>
#include <hip/hip_bf16.h>
#include <math.h>

typedef __attribute__((ext_vector_type(16))) _Float16 v16h;
typedef __attribute__((ext_vector_type(8)))  _Float16 v8h;
typedef __attribute__((ext_vector_type(8)))  float    v8f;

#define CCH   64
#define NUMC  5
#define HW    192
#define BATCH 16
#define XPAD  40   // halves per (row,w) slot: 80B stride, 16B-aligned fragments

// ---------------------------------------------------------------------------
// Stage 1a: A[n][i][j] = 0.5*(W[i][j][n] - W[j][i][n]),  W is (C,C,5)
// ---------------------------------------------------------------------------
__global__ void k_build_A(const float* __restrict__ W, float* __restrict__ A) {
  int idx = blockIdx.x * blockDim.x + threadIdx.x;   // 5*64*64 = 20480
  if (idx >= NUMC * CCH * CCH) return;
  int n = idx / (CCH * CCH);
  int r = idx - n * CCH * CCH;
  int i = r / CCH, j = r - i * CCH;
  float wij = W[(i * CCH + j) * NUMC + n];
  float wji = W[(j * CCH + i) * NUMC + n];
  A[idx] = 0.5f * (wij - wji);
}

// ---------------------------------------------------------------------------
// Stage 1b: 50 power iterations per slice, sigma, An = A / sigma
// ---------------------------------------------------------------------------
__device__ __forceinline__ float reduce64(float val, float* red, int t) {
  red[t] = val;
  __syncthreads();
#pragma unroll
  for (int s = 32; s > 0; s >>= 1) {
    if (t < s) red[t] += red[t + s];
    __syncthreads();
  }
  float r = red[0];
  __syncthreads();
  return r;
}

__global__ void k_power(const float* __restrict__ A, const float* __restrict__ u0,
                        const float* __restrict__ v0, float* __restrict__ An) {
  __shared__ float As[64 * 65];
  __shared__ float u[64], v[64], red[64];
  const int n = blockIdx.x, t = threadIdx.x;
  for (int i = 0; i < 64; ++i) As[i * 65 + t] = A[n * 4096 + i * 64 + t];
  float uv = u0[n * 64 + t];
  float vv = v0[n * 64 + t];
  __syncthreads();
  float s = reduce64(uv * uv, red, t);
  uv /= (sqrtf(s) + 1e-12f);
  s = reduce64(vv * vv, red, t);
  vv /= (sqrtf(s) + 1e-12f);
  u[t] = uv; v[t] = vv;
  __syncthreads();
  for (int it = 0; it < 50; ++it) {
    float a = 0.0f;
    for (int j = 0; j < 64; ++j) a += As[t * 65 + j] * u[j];       // (A u)_t
    s = reduce64(a * a, red, t);
    v[t] = a / (sqrtf(s) + 1e-12f);
    __syncthreads();
    float b = 0.0f;
    for (int i = 0; i < 64; ++i) b += As[i * 65 + t] * v[i];       // (A^T v)_t
    s = reduce64(b * b, red, t);
    u[t] = b / (sqrtf(s) + 1e-12f);
    __syncthreads();
  }
  float p = 0.0f;
  for (int j = 0; j < 64; ++j) p += As[t * 65 + j] * u[j];
  p *= v[t];
  float sigma = reduce64(p, red, t);
  for (int j = 0; j < 64; ++j) An[n * 4096 + t * 64 + j] = As[t * 65 + j] / sigma;
}

// ---------------------------------------------------------------------------
// Stage 1c: Fm = An + An^2/2! + ... + An^10/10! + I
// ---------------------------------------------------------------------------
__global__ void k_taylor(const float* __restrict__ An, float* __restrict__ Fm) {
  __shared__ float Am[4096], cur[4096];
  const int n = blockIdx.x, t = threadIdx.x;           // 256 threads
  for (int e = t; e < 4096; e += 256) {
    float a = An[n * 4096 + e];
    Am[e] = a; cur[e] = a;
  }
  __syncthreads();
  float fm[16], tmp[16];
#pragma unroll
  for (int q = 0; q < 16; ++q) fm[q] = Am[t * 16 + q];
  for (int i = 2; i <= 10; ++i) {
    float inv = 1.0f / (float)i;
#pragma unroll
    for (int q = 0; q < 16; ++q) {
      int e = t * 16 + q, row = e >> 6, col = e & 63;
      float s = 0.0f;
      for (int k = 0; k < 64; ++k) s += cur[(row << 6) + k] * Am[(k << 6) + col];
      tmp[q] = s * inv;
      fm[q] += tmp[q];
    }
    __syncthreads();
#pragma unroll
    for (int q = 0; q < 16; ++q) cur[t * 16 + q] = tmp[q];
    __syncthreads();
  }
#pragma unroll
  for (int q = 0; q < 16; ++q) {
    int e = t * 16 + q, row = e >> 6, col = e & 63;
    Fm[n * 4096 + e] = fm[q] + ((row == col) ? 1.0f : 0.0f);
  }
}

// ---------------------------------------------------------------------------
// Stage 1d: expand via IDX, 3x3 real ifft2 (cos table {1,-0.5,-0.5}), pack f16
// wg[o][k], k = (kh*3+kw)*64 + i
// ---------------------------------------------------------------------------
__global__ void k_wgemm(const float* __restrict__ Fm, _Float16* __restrict__ wg) {
  int idx = blockIdx.x * blockDim.x + threadIdx.x;     // 4096
  if (idx >= CCH * CCH) return;
  int o = idx / CCH, i = idx - o * CCH;
  const int IDXT[9] = {0, 1, 1, 2, 3, 4, 2, 4, 3};
  const float ctab[3] = {1.0f, -0.5f, -0.5f};
  float f[9];
#pragma unroll
  for (int t = 0; t < 9; ++t) f[t] = Fm[IDXT[t] * CCH * CCH + o * CCH + i];
#pragma unroll
  for (int kh = 0; kh < 3; ++kh)
#pragma unroll
    for (int kw = 0; kw < 3; ++kw) {
      float s = 0.0f;
#pragma unroll
      for (int j = 0; j < 3; ++j)
#pragma unroll
        for (int k2 = 0; k2 < 3; ++k2)
          s += f[j * 3 + k2] * ctab[(j * kh + k2 * kw) % 3];
      wg[o * 576 + (kh * 3 + kw) * CCH + i] = (_Float16)(s * (1.0f / 9.0f));
    }
}

// ---------------------------------------------------------------------------
// Stage 2: wrap-dilated conv as WMMA GEMM, one block per (n, h).
// M=64 out-ch, N=192 w-pixels, K=576 (9 taps x 64 ch), 16 waves.
// ---------------------------------------------------------------------------
__global__ __launch_bounds__(512, 1)
void k_conv(const float* __restrict__ x, const _Float16* __restrict__ wg,
            const float* __restrict__ bias, float* __restrict__ out) {
  const int nb   = blockIdx.x;
  const int n    = nb / HW;
  const int h    = nb - n * HW;
  const int tid  = threadIdx.x;
  const int lane = tid & 31;
  const int wave = tid >> 5;        // 0..15
  const int o_tile = wave & 3;      // 0..3
  const int pgrp   = wave >> 2;     // 0..3

  __shared__ __align__(16) _Float16 xs[3 * HW * XPAD];   // 46,080 B

  int hsrc[3];
  hsrc[0] = (h + 128) % HW;  // kh=0 -> h-64 mod 192
  hsrc[1] = h;               // kh=1
  hsrc[2] = (h + 64) % HW;   // kh=2

  const int nlo   = lane & 15;
  const int khalf = (lane & 16) >> 1;      // 0 or 8: half-wave K (and M) offset

  // precompute wrapped source columns for all (kw, j) once per lane
  int wsrc[3][3];
#pragma unroll
  for (int j = 0; j < 3; ++j) {
    int wcol = pgrp * 48 + j * 16 + nlo;
#pragma unroll
    for (int kw = 0; kw < 3; ++kw) {
      int t = wcol + ((kw == 0) ? 128 : (kw == 1) ? 0 : 64);
      wsrc[kw][j] = (t >= HW) ? (t - HW) : t;
    }
  }

  const int o_row = o_tile * 16 + nlo;
  const _Float16* wbase = wg + o_row * 576 + khalf;   // 16B aligned

  v8f acc[3] = {};

  for (int phase = 0; phase < 2; ++phase) {
    const int cbase = phase * 32;
    // stage 3 rows x 32 channels x 192 cols of x into LDS (f16), transposed
    for (int e = tid; e < 3 * 32 * HW; e += 512) {
      int rh  = e / (32 * HW);
      int rem = e - rh * (32 * HW);
      int ii  = rem / HW;
      int w   = rem - ii * HW;
      float xv = x[(((size_t)n * CCH + cbase + ii) * HW + hsrc[rh]) * HW + w];
      xs[(rh * HW + w) * XPAD + ii] = (_Float16)xv;
    }
    __syncthreads();

#pragma unroll
    for (int tap = 0; tap < 9; ++tap) {
      const int kh = tap / 3, kw = tap - kh * 3;      // compile-time
      // A fragment: two aligned b128 loads (L2-resident weights)
      const _Float16* ap = wbase + tap * 64 + cbase;
      v8h alo = *(const v8h*)ap;
      v8h ahi = *(const v8h*)(ap + 16);
      v16h a = __builtin_shufflevector(alo, ahi, 0, 1, 2, 3, 4, 5, 6, 7,
                                       8, 9, 10, 11, 12, 13, 14, 15);
#pragma unroll
      for (int j = 0; j < 3; ++j) {
        const _Float16* bp = xs + (kh * HW + wsrc[kw][j]) * XPAD + khalf;
        v8h blo = *(const v8h*)bp;
        v8h bhi = *(const v8h*)(bp + 16);
        v16h b = __builtin_shufflevector(blo, bhi, 0, 1, 2, 3, 4, 5, 6, 7,
                                         8, 9, 10, 11, 12, 13, 14, 15);
        acc[j] = __builtin_amdgcn_wmma_f32_16x16x32_f16(
            false, a, false, b, (short)0, acc[j], false, false);
      }
    }
    __syncthreads();   // protect xs before next phase's load
  }

  // D layout: VGPR r -> M = r + khalf; N = lane%16
  float bvals[8];
#pragma unroll
  for (int r = 0; r < 8; ++r) bvals[r] = bias[o_tile * 16 + r + khalf];
#pragma unroll
  for (int j = 0; j < 3; ++j) {
    int wcol = pgrp * 48 + j * 16 + nlo;
#pragma unroll
    for (int r = 0; r < 8; ++r) {
      int o = o_tile * 16 + r + khalf;
      out[(((size_t)n * CCH + o) * HW + h) * HW + wcol] = acc[j][r] + bvals[r];
    }
  }
}

// ---------------------------------------------------------------------------
extern "C" void kernel_launch(void* const* d_in, const int* in_sizes, int n_in,
                              void* d_out, int out_size, void* d_ws, size_t ws_size,
                              hipStream_t stream) {
  const float* x    = (const float*)d_in[0];   // (16,64,192,192)
  const float* W    = (const float*)d_in[1];   // (64,64,5)
  const float* bias = (const float*)d_in[2];   // (64,)
  const float* u0   = (const float*)d_in[3];   // (5,1,64)
  const float* v0   = (const float*)d_in[4];   // (5,64,1)
  float* out = (float*)d_out;

  char* ws = (char*)d_ws;
  float*    A  = (float*)(ws);                 //  81,920 B
  float*    An = (float*)(ws + 81920);         //  81,920 B
  float*    Fm = (float*)(ws + 163840);        //  81,920 B
  _Float16* wg = (_Float16*)(ws + 245760);     //  73,728 B

  k_build_A<<<80, 256, 0, stream>>>(W, A);
  k_power  <<<NUMC, 64, 0, stream>>>(A, u0, v0, An);
  k_taylor <<<NUMC, 256, 0, stream>>>(An, Fm);
  k_wgemm  <<<16, 256, 0, stream>>>(Fm, wg);
  k_conv   <<<BATCH * HW, 512, 0, stream>>>(x, wg, bias, out);
}
